// Working_Condition_Decomp_85718957294132
// MI455X (gfx1250) — matrix operand order repaired and checked by reading, not compiled, not run on hardware
//
#include <hip/hip_runtime.h>
#include <hip/hip_bf16.h>
#include <math.h>

// ---------------------------------------------------------------------------
// Problem constants (match reference: B=8, L=4096, D=512, H=4, E=128, TOP_K=8)
// ---------------------------------------------------------------------------
#define B_SZ   8
#define L_SEQ  4096
#define D_MOD  512
#define TOPK   8
#define KCONV  1536           // 3 * D

// WMMA GEMM tiling: block = 128x128, k-step 32, 256 threads = 8 waves (wave32)
#define BM 128
#define BN 128
#define BK 32
#define LDSR 40               // LDS row stride in bf16 elems (80B: 16B aligned, conflict-free)

typedef __bf16 bf16_t;
typedef bf16_t v16bf __attribute__((ext_vector_type(16)));
typedef bf16_t v8bf  __attribute__((ext_vector_type(8)));
typedef bf16_t v4bf  __attribute__((ext_vector_type(4)));
typedef float  v8f   __attribute__((ext_vector_type(8)));

// ---------------------------------------------------------------------------
// CDNA5 async global->LDS copy (ASYNCcnt-tracked, bypasses VGPRs).
// the LDS address comes from the VDST operand; generic __shared__ pointers carry the LDS
// byte offset in their low 32 bits, so truncation yields the VDST value.
// ---------------------------------------------------------------------------
__device__ __forceinline__ void async_b128(uint32_t lds_byte_off, const void* gaddr) {
  asm volatile("global_load_async_to_lds_b128 %0, %1, off"
               :: "v"(lds_byte_off), "v"(gaddr)
               : "memory");
}
__device__ __forceinline__ void wait_async0() {
#if __has_builtin(__builtin_amdgcn_s_wait_asynccnt)
  __builtin_amdgcn_s_wait_asynccnt(0);
#else
  asm volatile("s_wait_asynccnt 0" ::: "memory");
#endif
}
__device__ __forceinline__ uint32_t lds_off(const void* p) {
  return (uint32_t)(uintptr_t)p;
}

// ---------------------------------------------------------------------------
// Fragment loaders matching CDNA5 ISA wave32 WMMA register layouts (CDNA5 ISA)
// A 16x32 bf16: lane(0..15)=row M, VGPR0..3 K=0..7, VGPR4..7 K=16..23;
//               lanes 16..31 same rows, K offset +8 (K=8..15 / 24..31).
// B 32x16 bf16: lane(0..15)=col N, K=0..7 / 16..23; lanes 16..31: K=8..15 / 24..31.
// ---------------------------------------------------------------------------
__device__ __forceinline__ v16bf load_frag_a(const bf16_t* base, int lane) {
  const int row = lane & 15;
  const int kb  = (lane >> 4) << 3;               // 0 or 8
  const bf16_t* p = base + row * LDSR + kb;
  v8bf lo = *reinterpret_cast<const v8bf*>(p);         // K = kb .. kb+7
  v8bf hi = *reinterpret_cast<const v8bf*>(p + 16);    // K = kb+16 .. kb+23
  return __builtin_shufflevector(lo, hi, 0,1,2,3,4,5,6,7,8,9,10,11,12,13,14,15);
}

__device__ __forceinline__ v16bf load_frag_b(const bf16_t* base, int lane) {
  const int n  = lane & 15;
  const int kb = (lane >> 4) << 3;                // 0 or 8
  const bf16_t* p = base + n * LDSR + kb;
  v8bf lo = *reinterpret_cast<const v8bf*>(p);
  v8bf hi = *reinterpret_cast<const v8bf*>(p + 16);
  return __builtin_shufflevector(lo, hi, 0,1,2,3,4,5,6,7,8,9,10,11,12,13,14,15);
}

// 4x2 subtile WMMA update from the two LDS tile bases.
__device__ __forceinline__ void wmma_tile(
    v8f (&acc)[4][2], const bf16_t* baseA, const bf16_t* baseB,
    int wm, int wn, int lane)
{
  v16bf af[4], bfr[2];
#pragma unroll
  for (int i = 0; i < 4; ++i) af[i]  = load_frag_a(baseA + (wm * 64 + i * 16) * LDSR, lane);
#pragma unroll
  for (int j = 0; j < 2; ++j) bfr[j] = load_frag_b(baseB + (wn * 32 + j * 16) * LDSR, lane);
#pragma unroll
  for (int i = 0; i < 4; ++i)
#pragma unroll
    for (int j = 0; j < 2; ++j)
      acc[i][j] = __builtin_amdgcn_wmma_f32_16x16x32_bf16(
          false, af[i], false, bfr[j], (short)0, acc[i][j], false, false);
}

// ---------------------------------------------------------------------------
// Generic GEMM: C[M,N] = A[M,K] * W[N,K]^T (+bias) (+residual), K = KDIM.
// Double-buffered async global->LDS pipeline, branch-free steady state:
// last (no-prefetch) step peeled; buffers swapped via pointer rotation and
// the steady loop fully unrolled (compile-time trip count).
// FINAL=false : write bf16; FINAL=true : write fp32 (val + resid).
// ---------------------------------------------------------------------------
template<bool FINAL, int KDIM>
__global__ __launch_bounds__(256) void gemm_wmma(
    const bf16_t* __restrict__ A, const bf16_t* __restrict__ Bw,
    const float* __restrict__ bias, const float* __restrict__ resid,
    void* __restrict__ Cout, int N)
{
  __shared__ bf16_t lsA[2][BM * LDSR];
  __shared__ bf16_t lsB[2][BN * LDSR];

  const int m0 = blockIdx.x * BM;
  const int n0 = blockIdx.y * BN;
  const int tid  = threadIdx.x;
  const int lane = tid & 31;
  const int wm   = (tid >> 5) >> 2;     // 0..1 : 64 rows each
  const int wn   = (tid >> 5) & 3;      // 0..3 : 32 cols each
  const int ldrow = tid >> 1;           // 0..127
  const int ldcol = (tid & 1) << 4;     // 0 or 16

  uint32_t laC = lds_off(&lsA[0][ldrow * LDSR + ldcol]);
  uint32_t laN = lds_off(&lsA[1][ldrow * LDSR + ldcol]);
  uint32_t lbC = lds_off(&lsB[0][ldrow * LDSR + ldcol]);
  uint32_t lbN = lds_off(&lsB[1][ldrow * LDSR + ldcol]);
  const bf16_t* baC = lsA[0]; const bf16_t* baN = lsA[1];
  const bf16_t* bbC = lsB[0]; const bf16_t* bbN = lsB[1];

  const bf16_t* gA = A  + (size_t)(m0 + ldrow) * KDIM + ldcol;
  const bf16_t* gB = Bw + (size_t)(n0 + ldrow) * KDIM + ldcol;

  v8f acc[4][2] = {};

  // prologue: async-load tile 0 into buffer 0
  async_b128(laC, gA);      async_b128(laC + 16, gA + 8);
  async_b128(lbC, gB);      async_b128(lbC + 16, gB + 8);

  constexpr int NSTEP = KDIM / BK;
#pragma unroll
  for (int t = 0; t < NSTEP - 1; ++t) {
    wait_async0();
    __syncthreads();                       // tile[cur] ready; prior reads done
    const bf16_t* pa = gA + (t + 1) * BK;  // unconditional prefetch of next tile
    const bf16_t* pb = gB + (t + 1) * BK;
    async_b128(laN, pa);  async_b128(laN + 16, pa + 8);
    async_b128(lbN, pb);  async_b128(lbN + 16, pb + 8);
    wmma_tile(acc, baC, bbC, wm, wn, lane);
    uint32_t tu;  const bf16_t* tp;
    tu = laC; laC = laN; laN = tu;   tu = lbC; lbC = lbN; lbN = tu;
    tp = baC; baC = baN; baN = tp;   tp = bbC; bbC = bbN; bbN = tp;
  }
  wait_async0();
  __syncthreads();
  wmma_tile(acc, baC, bbC, wm, wn, lane);  // final tile, no prefetch

  // C 16x16 f32 layout: lanes 0-15 -> N=lane, M=VGPRr; lanes 16-31 -> M=8+r
  const int cn = lane & 15;
  const int mb = (lane >> 4) << 3;
#pragma unroll
  for (int i = 0; i < 4; ++i) {
#pragma unroll
    for (int j = 0; j < 2; ++j) {
      const int gm = m0 + wm * 64 + i * 16 + mb;
      const int gn = n0 + wn * 32 + j * 16 + cn;
      const float bv = bias ? bias[gn] : 0.0f;
#pragma unroll
      for (int r = 0; r < 8; ++r) {
        const size_t off = (size_t)(gm + r) * N + gn;
        const float val = acc[i][j][r] + bv;
        if (FINAL) reinterpret_cast<float*>(Cout)[off] = val + resid[off];
        else       reinterpret_cast<bf16_t*>(Cout)[off] = (bf16_t)val;
      }
    }
  }
}

// ---------------------------------------------------------------------------
// Fused Gram + diagonal reduction (replaces the reference FFT):
// mean_value[b,tau] += (1/D) * sum_t <q[b,t,:], k[b,(t-tau)%L,:]>
// One block = 128x128 tile of Q*K^T; its 255 diagonals reduce in LDS, then
// atomically accumulate into mean_value[b, tau].
// ---------------------------------------------------------------------------
__global__ __launch_bounds__(256) void gram_corr(
    const bf16_t* __restrict__ Qm, const bf16_t* __restrict__ Km,
    float* __restrict__ meanv)
{
  __shared__ bf16_t lsA[2][BM * LDSR];
  __shared__ bf16_t lsB[2][BN * LDSR];
  __shared__ float  corr[2 * BM - 1];   // 255 diagonals per tile

  const int b  = blockIdx.y;
  const int t0 = (blockIdx.x >> 5) * BM;
  const int s0 = (blockIdx.x & 31) * BN;

  const int tid  = threadIdx.x;
  const int lane = tid & 31;
  const int wm   = (tid >> 5) >> 2;
  const int wn   = (tid >> 5) & 3;
  const int ldrow = tid >> 1;
  const int ldcol = (tid & 1) << 4;

  uint32_t laC = lds_off(&lsA[0][ldrow * LDSR + ldcol]);
  uint32_t laN = lds_off(&lsA[1][ldrow * LDSR + ldcol]);
  uint32_t lbC = lds_off(&lsB[0][ldrow * LDSR + ldcol]);
  uint32_t lbN = lds_off(&lsB[1][ldrow * LDSR + ldcol]);
  const bf16_t* baC = lsA[0]; const bf16_t* baN = lsA[1];
  const bf16_t* bbC = lsB[0]; const bf16_t* bbN = lsB[1];

  const bf16_t* gA = Qm + (size_t)b * L_SEQ * D_MOD + (size_t)(t0 + ldrow) * D_MOD + ldcol;
  const bf16_t* gB = Km + (size_t)b * L_SEQ * D_MOD + (size_t)(s0 + ldrow) * D_MOD + ldcol;

  v8f acc[4][2] = {};

  async_b128(laC, gA);      async_b128(laC + 16, gA + 8);
  async_b128(lbC, gB);      async_b128(lbC + 16, gB + 8);

  constexpr int NSTEP = D_MOD / BK;
#pragma unroll
  for (int t = 0; t < NSTEP - 1; ++t) {
    wait_async0();
    __syncthreads();
    const bf16_t* pa = gA + (t + 1) * BK;
    const bf16_t* pb = gB + (t + 1) * BK;
    async_b128(laN, pa);  async_b128(laN + 16, pa + 8);
    async_b128(lbN, pb);  async_b128(lbN + 16, pb + 8);
    wmma_tile(acc, baC, bbC, wm, wn, lane);
    uint32_t tu;  const bf16_t* tp;
    tu = laC; laC = laN; laN = tu;   tu = lbC; lbC = lbN; lbN = tu;
    tp = baC; baC = baN; baN = tp;   tp = bbC; bbC = bbN; bbN = tp;
  }
  wait_async0();
  __syncthreads();
  wmma_tile(acc, baC, bbC, wm, wn, lane);

  for (int i = tid; i < 2 * BM - 1; i += 256) corr[i] = 0.0f;
  __syncthreads();

  const int cn = lane & 15;
  const int mb = (lane >> 4) << 3;
#pragma unroll
  for (int i = 0; i < 4; ++i)
#pragma unroll
    for (int j = 0; j < 2; ++j) {
      const int sn = wn * 32 + j * 16 + cn;
#pragma unroll
      for (int r = 0; r < 8; ++r) {
        const int tm = wm * 64 + i * 16 + mb + r;
        atomicAdd(&corr[tm - sn + (BM - 1)], acc[i][j][r]);
      }
    }
  __syncthreads();

  for (int i = tid; i < 2 * BM - 1; i += 256) {
    const int tau = (t0 - s0 + i - (BM - 1)) & (L_SEQ - 1);
    atomicAdd(&meanv[b * L_SEQ + tau], corr[i] * (1.0f / (float)D_MOD));
  }
}

// ---------------------------------------------------------------------------
// Circular conv1d as GEMM: y[b,l,o] = sum_{kk,c} x_w[b,(l+kk-1)%L,c]*Wc[o,kk*512+c]
// K = 1536; the A loader applies a circular row shift per kk segment.
// ---------------------------------------------------------------------------
__global__ __launch_bounds__(256) void conv_gemm_wmma(
    const bf16_t* __restrict__ Axw, const bf16_t* __restrict__ Bw,
    float* __restrict__ Cout)
{
  __shared__ bf16_t lsA[2][BM * LDSR];
  __shared__ bf16_t lsB[2][BN * LDSR];

  const int m0 = blockIdx.x * BM;       // rows over B*L (128-row block stays in one batch)
  const int n0 = blockIdx.y * BN;
  const int tid  = threadIdx.x;
  const int lane = tid & 31;
  const int wm   = (tid >> 5) >> 2;
  const int wn   = (tid >> 5) & 3;
  const int ldrow = tid >> 1;
  const int ldcol = (tid & 1) << 4;

  uint32_t laC = lds_off(&lsA[0][ldrow * LDSR + ldcol]);
  uint32_t laN = lds_off(&lsA[1][ldrow * LDSR + ldcol]);
  uint32_t lbC = lds_off(&lsB[0][ldrow * LDSR + ldcol]);
  uint32_t lbN = lds_off(&lsB[1][ldrow * LDSR + ldcol]);
  const bf16_t* baC = lsA[0]; const bf16_t* baN = lsA[1];
  const bf16_t* bbC = lsB[0]; const bf16_t* bbN = lsB[1];

  const int grow = m0 + ldrow;
  const int bb = grow >> 12;
  const int l  = grow & (L_SEQ - 1);

  auto a_ptr = [&](int k0) -> const bf16_t* {
    const int kk = k0 >> 9;                       // conv tap 0..2
    const int c0 = k0 & (D_MOD - 1);
    const int sl = (l + kk - 1) & (L_SEQ - 1);    // circular padding
    return Axw + ((size_t)((bb << 12) + sl) * D_MOD + c0 + ldcol);
  };
  const bf16_t* gB = Bw + (size_t)(n0 + ldrow) * KCONV + ldcol;

  v8f acc[4][2] = {};

  {
    const bf16_t* pa = a_ptr(0);
    async_b128(laC, pa);      async_b128(laC + 16, pa + 8);
    async_b128(lbC, gB);      async_b128(lbC + 16, gB + 8);
  }
  constexpr int NSTEP = KCONV / BK;     // 48
#pragma unroll 4
  for (int t = 0; t < NSTEP - 1; ++t) {
    wait_async0();
    __syncthreads();
    const bf16_t* pa = a_ptr((t + 1) * BK);
    const bf16_t* pb = gB + (t + 1) * BK;
    async_b128(laN, pa);  async_b128(laN + 16, pa + 8);
    async_b128(lbN, pb);  async_b128(lbN + 16, pb + 8);
    wmma_tile(acc, baC, bbC, wm, wn, lane);
    uint32_t tu;  const bf16_t* tp;
    tu = laC; laC = laN; laN = tu;   tu = lbC; lbC = lbN; lbN = tu;
    tp = baC; baC = baN; baN = tp;   tp = bbC; bbC = bbN; bbN = tp;
  }
  wait_async0();
  __syncthreads();
  wmma_tile(acc, baC, bbC, wm, wn, lane);

  const int cn = lane & 15;
  const int mb = (lane >> 4) << 3;
#pragma unroll
  for (int i = 0; i < 4; ++i)
#pragma unroll
    for (int j = 0; j < 2; ++j) {
      const int gm = m0 + wm * 64 + i * 16 + mb;
      const int gn = n0 + wn * 32 + j * 16 + cn;
#pragma unroll
      for (int r = 0; r < 8; ++r)
        Cout[(size_t)(gm + r) * D_MOD + gn] = acc[i][j][r];
    }
}

// ---------------------------------------------------------------------------
// Small helper kernels (16B-vectorized: these are pure-bandwidth steps)
// ---------------------------------------------------------------------------
__global__ void f32_to_bf16_v4(const float* __restrict__ in, bf16_t* __restrict__ out, int n4) {
  for (int i = blockIdx.x * blockDim.x + threadIdx.x; i < n4; i += gridDim.x * blockDim.x) {
    const float4 f = reinterpret_cast<const float4*>(in)[i];
    v4bf o;
    o[0] = (bf16_t)f.x; o[1] = (bf16_t)f.y; o[2] = (bf16_t)f.z; o[3] = (bf16_t)f.w;
    reinterpret_cast<v4bf*>(out)[i] = o;
  }
}

// Wconv[o,c,kk] (o*1536 + c*3 + kk)  ->  Wcb[o, kk*512 + c]
__global__ void conv_w_relayout(const float* __restrict__ w, bf16_t* __restrict__ out) {
  const int i = blockIdx.x * blockDim.x + threadIdx.x;   // 512*512*3 total
  const int o  = i / KCONV;
  const int r  = i - o * KCONV;
  const int kk = r >> 9;
  const int c  = r & (D_MOD - 1);
  out[i] = (bf16_t)w[(o * D_MOD + c) * 3 + kk];
}

// Top-8 + softmax per batch (1 workgroup per batch).
__global__ __launch_bounds__(256) void topk_softmax(
    const float* __restrict__ meanv, int* __restrict__ delays, float* __restrict__ wts)
{
  __shared__ float vals[L_SEQ];
  __shared__ float rv[256];
  __shared__ int   ri[256];
  __shared__ float topv[TOPK];
  __shared__ int   topi[TOPK];
  const int b = blockIdx.x, tid = threadIdx.x;
  for (int i = tid; i < L_SEQ; i += 256) vals[i] = meanv[b * L_SEQ + i];
  __syncthreads();
  for (int it = 0; it < TOPK; ++it) {
    float best = -INFINITY; int bi = 0;
    for (int i = tid; i < L_SEQ; i += 256) {
      const float v = vals[i];
      if (v > best) { best = v; bi = i; }
    }
    rv[tid] = best; ri[tid] = bi;
    __syncthreads();
    for (int s = 128; s > 0; s >>= 1) {
      if (tid < s && rv[tid + s] > rv[tid]) { rv[tid] = rv[tid + s]; ri[tid] = ri[tid + s]; }
      __syncthreads();
    }
    if (tid == 0) { topv[it] = rv[0]; topi[it] = ri[0]; vals[ri[0]] = -INFINITY; }
    __syncthreads();
  }
  if (tid == 0) {
    float mx = topv[0];
    for (int i = 1; i < TOPK; ++i) mx = fmaxf(mx, topv[i]);
    float e[TOPK], s = 0.0f;
    for (int i = 0; i < TOPK; ++i) { e[i] = __expf(topv[i] - mx); s += e[i]; }
    for (int i = 0; i < TOPK; ++i) { wts[b * TOPK + i] = e[i] / s; delays[b * TOPK + i] = topi[i]; }
  }
}

// agg[b,l,d] = sum_i w[b,i] * v[b,(l+delay_i)%L,d]
// 256 threads = 4 rows x 64 chunks of 8 bf16 (16B vector gathers).
__global__ __launch_bounds__(256) void aggregate(
    const bf16_t* __restrict__ vb, const int* __restrict__ delays,
    const float* __restrict__ wts, bf16_t* __restrict__ aggb)
{
  const int row = blockIdx.x * 4 + (threadIdx.x >> 6);   // over B*L
  const int b = row >> 12;
  const int l = row & (L_SEQ - 1);
  const int c = (threadIdx.x & 63) * 8;
  __shared__ int   dl[TOPK];
  __shared__ float w[TOPK];
  if (threadIdx.x < TOPK) {
    dl[threadIdx.x] = delays[b * TOPK + threadIdx.x];
    w[threadIdx.x]  = wts[b * TOPK + threadIdx.x];
  }
  __syncthreads();
  const size_t base = (size_t)b * L_SEQ * D_MOD;
  float acc[8] = {};
#pragma unroll
  for (int i = 0; i < TOPK; ++i) {
    const int sl = (l + dl[i]) & (L_SEQ - 1);
    const v8bf vv = *reinterpret_cast<const v8bf*>(vb + base + (size_t)sl * D_MOD + c);
    const float wi = w[i];
#pragma unroll
    for (int j = 0; j < 8; ++j) acc[j] += wi * (float)vv[j];
  }
  v8bf o;
#pragma unroll
  for (int j = 0; j < 8; ++j) o[j] = (bf16_t)acc[j];
  *reinterpret_cast<v8bf*>(aggb + base + (size_t)l * D_MOD + c) = o;
}

// ---------------------------------------------------------------------------
// Host launcher
// ---------------------------------------------------------------------------
extern "C" void kernel_launch(void* const* d_in, const int* in_sizes, int n_in,
                              void* d_out, int out_size, void* d_ws, size_t ws_size,
                              hipStream_t stream)
{
  const float* x_s   = (const float*)d_in[0];
  const float* x_w   = (const float*)d_in[1];
  const float* Wq    = (const float*)d_in[2];
  const float* bq    = (const float*)d_in[3];
  const float* Wk    = (const float*)d_in[4];
  const float* bk    = (const float*)d_in[5];
  const float* Wv    = (const float*)d_in[6];
  const float* bv    = (const float*)d_in[7];
  const float* Wo    = (const float*)d_in[8];
  const float* bo    = (const float*)d_in[9];
  const float* Wconv = (const float*)d_in[10];

  const size_t BLD = (size_t)B_SZ * L_SEQ * D_MOD;   // 16,777,216
  char* ws = (char*)d_ws;
  size_t off = 0;
  auto walloc = [&](size_t bytes) -> void* {
    void* p = ws + off;
    off += (bytes + 255) & ~(size_t)255;
    return p;
  };

  bf16_t* xsb  = (bf16_t*)walloc(BLD * 2);
  bf16_t* xwb  = (bf16_t*)walloc(BLD * 2);
  bf16_t* qb   = (bf16_t*)walloc(BLD * 2);
  bf16_t* kb   = (bf16_t*)walloc(BLD * 2);
  bf16_t* vbuf = (bf16_t*)walloc(BLD * 2);
  bf16_t* aggb = (bf16_t*)walloc(BLD * 2);
  bf16_t* Wqb  = (bf16_t*)walloc((size_t)D_MOD * D_MOD * 2);
  bf16_t* Wkb  = (bf16_t*)walloc((size_t)D_MOD * D_MOD * 2);
  bf16_t* Wvb  = (bf16_t*)walloc((size_t)D_MOD * D_MOD * 2);
  bf16_t* Wob  = (bf16_t*)walloc((size_t)D_MOD * D_MOD * 2);
  bf16_t* Wcb  = (bf16_t*)walloc((size_t)D_MOD * KCONV * 2);
  float*  meanv  = (float*)walloc((size_t)B_SZ * L_SEQ * 4);
  int*    delays = (int*)walloc(B_SZ * TOPK * 4);
  float*  wts    = (float*)walloc(B_SZ * TOPK * 4);

  hipMemsetAsync(meanv, 0, (size_t)B_SZ * L_SEQ * 4, stream);

  // fp32 -> bf16 conversions / relayout
  f32_to_bf16_v4<<<2048, 256, 0, stream>>>(x_s, xsb, (int)(BLD / 4));
  f32_to_bf16_v4<<<2048, 256, 0, stream>>>(x_w, xwb, (int)(BLD / 4));
  f32_to_bf16_v4<<<256, 256, 0, stream>>>(Wq, Wqb, D_MOD * D_MOD / 4);
  f32_to_bf16_v4<<<256, 256, 0, stream>>>(Wk, Wkb, D_MOD * D_MOD / 4);
  f32_to_bf16_v4<<<256, 256, 0, stream>>>(Wv, Wvb, D_MOD * D_MOD / 4);
  f32_to_bf16_v4<<<256, 256, 0, stream>>>(Wo, Wob, D_MOD * D_MOD / 4);
  conv_w_relayout<<<(D_MOD * KCONV) / 256, 256, 0, stream>>>(Wconv, Wcb);

  const int M = B_SZ * L_SEQ;               // 32768
  const dim3 gP(M / BM, D_MOD / BN);        // (256, 4)

  // q from x_w; k,v from x_s
  gemm_wmma<false, D_MOD><<<gP, 256, 0, stream>>>(xwb, Wqb, bq, nullptr, qb,   D_MOD);
  gemm_wmma<false, D_MOD><<<gP, 256, 0, stream>>>(xsb, Wkb, bk, nullptr, kb,   D_MOD);
  gemm_wmma<false, D_MOD><<<gP, 256, 0, stream>>>(xsb, Wvb, bv, nullptr, vbuf, D_MOD);

  // autocorrelation mean_value via fused Gram-diagonal reduction
  gram_corr<<<dim3((L_SEQ / BM) * (L_SEQ / BN), B_SZ), 256, 0, stream>>>(qb, kb, meanv);

  // top-8 delays + softmax weights, then weighted circular aggregation of v
  topk_softmax<<<B_SZ, 256, 0, stream>>>(meanv, delays, wts);
  aggregate<<<(B_SZ * L_SEQ) / 4, 256, 0, stream>>>(vbuf, delays, wts, aggb);

  // x_s_out = x_s + agg @ Wo^T + bo   -> d_out[0 .. BLD)
  gemm_wmma<true, D_MOD><<<gP, 256, 0, stream>>>(aggb, Wob, bo, x_s, (float*)d_out, D_MOD);

  // x_w_out = circular conv1d(x_w)    -> d_out[BLD .. 2*BLD)
  conv_gemm_wmma<<<gP, 256, 0, stream>>>(xwb, Wcb, (float*)d_out + BLD);
}
